// SpatialTemporalMask_82145544504097
// MI455X (gfx1250) — compile-verified, matrix operand order checked by reading, not run on hardware
//
#include <hip/hip_runtime.h>
#include <hip/hip_bf16.h>
#include <math.h>

// Problem constants (fixed by the reference).
#define B_SZ 128
#define L_SZ 512
#define NTHREADS 256            // 8 wave32 waves per block
#define NWAVES (NTHREADS / 32)

#define MAXINTER_T 86400.0f
#define MAXINTER_D 50000.0f
#define EARTH_M    6367000.0f
#define EPS_N      1e-12f
#define DEG2RAD    0.017453292519943295f

typedef float    v2f  __attribute__((ext_vector_type(2)));
typedef float    v8f  __attribute__((ext_vector_type(8)));
typedef _Float16 v16h __attribute__((ext_vector_type(16)));

// ---------------------------------------------------------------------------
// Full-wave (32-lane) sum on the matrix pipe (confirmed lowering to
// v_wmma_f32_16x16x4_f32 in round 1).
//
// A is 16x4 f32 (2 VGPRs). Lane value in A-VGPR0, zero in A-VGPR1 gives
// A[m][0]=v_m (lanes 0-15), A[m][2]=v_{m+16} (lanes 16-31). With B = ones,
// D[m][n] = v_m + v_{m+16} for every n. Summing the 8 D components per lane
// covers rows 0-7 (lanes 0-15) / 8-15 (lanes 16-31); one xor-16 shuffle
// completes the 32-lane sum, all in f32.
// ---------------------------------------------------------------------------
__device__ __forceinline__ float wave_sum_wmma(float v) {
#if __has_builtin(__builtin_amdgcn_wmma_f32_16x16x4_f32)
  v2f a; a[0] = v;    a[1] = 0.0f;
  v2f b; b[0] = 1.0f; b[1] = 1.0f;
  v8f c = {};
  v8f d = __builtin_amdgcn_wmma_f32_16x16x4_f32(
      false, a, false, b, (short)0, c, false, false);
#else
  v16h a = {};
  a[0] = (_Float16)v;
  v16h b;
#pragma unroll
  for (int t = 0; t < 16; ++t) b[t] = (_Float16)1.0f;
  v8f c = {};
  v8f d = __builtin_amdgcn_wmma_f32_16x16x32_f16(
      false, a, false, b, (short)0, c, false, false);
#endif
  float s = d[0] + d[1] + d[2] + d[3] + d[4] + d[5] + d[6] + d[7];
  s += __shfl_xor(s, 16, 32);
  return s;
}

// ---------------------------------------------------------------------------
// Async staging: GLOBAL_LOAD_ASYNC_TO_LDS_B128 (gfx1250 TDM-class path).
// GVS addressing: mem = SGPR base + VGPR byte offset. The LDS destination
// address is the low 32 bits of the generic pointer (wave-relative LDS
// offset — the same mapping the async unit applies). Tracked on ASYNCcnt.
// ---------------------------------------------------------------------------
__device__ __forceinline__ void stage_one_async(const float* __restrict__ g,
                                                float* l, int tid) {
  // 512 floats = 128 x b128; waves 4-7 simply skip (whole-wave EXEC skip).
  for (int k4 = tid; k4 < L_SZ / 4; k4 += NTHREADS) {
    unsigned laddr = (unsigned)(uintptr_t)(l + 4 * k4);
    unsigned goff  = (unsigned)(k4 * 16);
    asm volatile("global_load_async_to_lds_b128 %0, %1, %2"
                 :: "v"(laddr), "v"(goff), "s"(g)
                 : "memory");
  }
}

__device__ __forceinline__ void async_wait_and_barrier() {
  asm volatile("s_wait_asynccnt 0" ::: "memory");
  __syncthreads();
}

// Masked |dt| and haversine distance for one (i,j) pair.
// sin^2(x/2) = (1-cos x)/2 and cos(a-b) = ca*cb + sa*sb, so no per-pair trig
// evals are needed (only sqrt + asinf).
__device__ __forceinline__ void pair_vals(
    float ti, float sli, float cli, float sloi, float cloi,
    float tj, float slj, float clj, float sloj, float cloj,
    bool m, float& tval, float& sval) {
  float dt = fminf(fabsf(ti - tj), MAXINTER_T);
  float cc = cli * clj;
  float cosdlat = fmaf(sli, slj, cc);
  float cosdlon = fmaf(sloi, sloj, cloi * cloj);
  float a = fmaf(cc, 0.5f * (1.0f - cosdlon), 0.5f * (1.0f - cosdlat));
  a = fminf(fmaxf(a, 0.0f), 1.0f);
  float s = fminf(2.0f * asinf(sqrtf(a)) * EARTH_M, MAXINTER_D);
  tval = m ? dt : 0.0f;
  sval = m ? s : 0.0f;
}

// ---------------------------------------------------------------------------
// Kernel 0: per-point trig precompute (sin/cos of lat/lon in radians) and
// per-batch running-max init. Tiny (1.5 MB workspace, lives in L2 after).
// ---------------------------------------------------------------------------
__global__ void k_pre(const float* __restrict__ s_s, float* __restrict__ trig,
                      unsigned int* __restrict__ rmax_bits, int n) {
  const int idx = blockIdx.x * blockDim.x + threadIdx.x;
  if (idx < B_SZ) rmax_bits[idx] = 0u;  // +0.0f bits; r >= 0 always
  if (idx >= n) return;
  const float lon = s_s[2 * idx]     * DEG2RAD;
  const float lat = s_s[2 * idx + 1] * DEG2RAD;
  float sl, cl, so, co;
  sincosf(lat, &sl, &cl);
  sincosf(lon, &so, &co);
  trig[0 * n + idx] = sl;
  trig[1 * n + idx] = cl;
  trig[2 * n + idx] = so;
  trig[3 * n + idx] = co;
}

// ---------------------------------------------------------------------------
// Kernel 1: one block per (batch, row). Computes the masked t/s rows once
// (cached in LDS), row sum-of-squares via WMMA-assisted reductions, writes
// 1/max(||row||,eps), then the row's max of r = t/||t|| + s/||s|| and folds
// it into the per-batch global max with a bitwise atomicMax.
// ---------------------------------------------------------------------------
__global__ __launch_bounds__(NTHREADS) void k_norms_rmax(
    const float* __restrict__ t_s, const int* __restrict__ val_len,
    const float* __restrict__ trig,
    float* __restrict__ inv_nt, float* __restrict__ inv_ns,
    unsigned int* __restrict__ rmax_bits) {
  const int i = blockIdx.x;
  const int b = blockIdx.y;
  const int tid = threadIdx.x;
  const int wid = tid >> 5, lane = tid & 31;
  const size_t base = (size_t)b * L_SZ;
  const int N = B_SZ * L_SZ;

  __shared__ alignas(16) float rows[5 * L_SZ];  // t, sinlat, coslat, sinlon, coslon
  __shared__ float tmat[L_SZ], smat[L_SZ];
  __shared__ float redA[NWAVES], redB[NWAVES], redM[NWAVES];
  float* t_row = rows + 0 * L_SZ;
  float* s_la  = rows + 1 * L_SZ;
  float* c_la  = rows + 2 * L_SZ;
  float* s_lo  = rows + 3 * L_SZ;
  float* c_lo  = rows + 4 * L_SZ;

  stage_one_async(t_s + base,          t_row, tid);
  stage_one_async(trig + 0 * N + base, s_la,  tid);
  stage_one_async(trig + 1 * N + base, c_la,  tid);
  stage_one_async(trig + 2 * N + base, s_lo,  tid);
  stage_one_async(trig + 3 * N + base, c_lo,  tid);
  async_wait_and_barrier();

  const int vl = val_len[b];
  const bool row_in = (i < vl);
  const float ti = t_row[i], sli = s_la[i], cli = c_la[i];
  const float sloi = s_lo[i], cloi = c_lo[i];

  float ssqt = 0.0f, ssqs = 0.0f;
  for (int j = tid; j < L_SZ; j += NTHREADS) {
    const bool m = row_in && (j <= i) && (j < vl);
    float tv, sv;
    pair_vals(ti, sli, cli, sloi, cloi,
              t_row[j], s_la[j], c_la[j], s_lo[j], c_lo[j], m, tv, sv);
    tmat[j] = tv;
    smat[j] = sv;
    ssqt = fmaf(tv, tv, ssqt);
    ssqs = fmaf(sv, sv, ssqs);
  }

  // Block reduction: WMMA wave sum (2^-33 scaling is exact in f32 and keeps
  // the f16 fallback in range), then 8-wave combine via LDS.
  float wt = wave_sum_wmma(ssqt * 0x1p-33f);
  float ws = wave_sum_wmma(ssqs * 0x1p-33f);
  if (lane == 0) { redA[wid] = wt; redB[wid] = ws; }
  __syncthreads();
  float tt = 0.0f, ts = 0.0f;
#pragma unroll
  for (int w = 0; w < NWAVES; ++w) { tt += redA[w]; ts += redB[w]; }
  tt *= 0x1p33f;
  ts *= 0x1p33f;
  const float invt = 1.0f / fmaxf(sqrtf(tt), EPS_N);
  const float invs = 1.0f / fmaxf(sqrtf(ts), EPS_N);
  if (tid == 0) { inv_nt[base + i] = invt; inv_ns[base + i] = invs; }

  // Row max of r (zeros outside the mask match the reference's full-matrix max).
  float mx = 0.0f;
  for (int j = tid; j < L_SZ; j += NTHREADS)
    mx = fmaxf(mx, fmaf(tmat[j], invt, smat[j] * invs));
#pragma unroll
  for (int off = 16; off >= 1; off >>= 1)
    mx = fmaxf(mx, __shfl_xor(mx, off, 32));
  if (lane == 0) redM[wid] = mx;
  __syncthreads();
  if (tid == 0) {
    float m2 = redM[0];
#pragma unroll
    for (int w = 1; w < NWAVES; ++w) m2 = fmaxf(m2, redM[w]);
    atomicMax(rmax_bits + b, __float_as_uint(m2));  // non-negative floats
  }
}

// ---------------------------------------------------------------------------
// Kernel 2: one block per (batch, row). Recomputes r once (haversine total =
// 2 evals/pair across the whole pipeline), builds exp((rmax - r)*mask) with
// the masked-out exp(0)=1 terms included (matches softmax over the full row),
// WMMA-reduces the denominator, and streams the 134 MB output coalesced.
// z = rmax - r is in [0, ~2], so no max-shift is needed for exp stability.
// ---------------------------------------------------------------------------
__global__ __launch_bounds__(NTHREADS) void k_softmax_out(
    const float* __restrict__ t_s, const int* __restrict__ val_len,
    const float* __restrict__ trig,
    const float* __restrict__ inv_nt, const float* __restrict__ inv_ns,
    const unsigned int* __restrict__ rmax_bits,
    float* __restrict__ out) {
  const int i = blockIdx.x;
  const int b = blockIdx.y;
  const int tid = threadIdx.x;
  const int wid = tid >> 5, lane = tid & 31;
  const size_t base = (size_t)b * L_SZ;
  const int N = B_SZ * L_SZ;

  __shared__ alignas(16) float rows[5 * L_SZ];
  __shared__ float erow[L_SZ];
  __shared__ float redD[NWAVES];
  float* t_row = rows + 0 * L_SZ;
  float* s_la  = rows + 1 * L_SZ;
  float* c_la  = rows + 2 * L_SZ;
  float* s_lo  = rows + 3 * L_SZ;
  float* c_lo  = rows + 4 * L_SZ;

  stage_one_async(t_s + base,          t_row, tid);
  stage_one_async(trig + 0 * N + base, s_la,  tid);
  stage_one_async(trig + 1 * N + base, c_la,  tid);
  stage_one_async(trig + 2 * N + base, s_lo,  tid);
  stage_one_async(trig + 3 * N + base, c_lo,  tid);
  async_wait_and_barrier();

  const int vl = val_len[b];
  const bool row_in = (i < vl);
  const float rmax = __uint_as_float(rmax_bits[b]);
  const float invt = inv_nt[base + i];
  const float invs = inv_ns[base + i];
  const float ti = t_row[i], sli = s_la[i], cli = c_la[i];
  const float sloi = s_lo[i], cloi = c_lo[i];

  float acc = 0.0f;
  for (int j = tid; j < L_SZ; j += NTHREADS) {
    const bool m = row_in && (j <= i) && (j < vl);
    float tv, sv;
    pair_vals(ti, sli, cli, sloi, cloi,
              t_row[j], s_la[j], c_la[j], s_lo[j], c_lo[j], m, tv, sv);
    const float r = fmaf(tv, invt, sv * invs);
    const float z = m ? (rmax - r) : 0.0f;  // masked-out -> exp(0) = 1
    const float e = expf(z);
    erow[j] = e;
    acc += e;
  }

  // Denominator: acc <= 2*e^2 ~ 14.8; /16 keeps the f16 fallback in range.
  float wsum = wave_sum_wmma(acc * 0.0625f);
  if (lane == 0) redD[wid] = wsum;
  __syncthreads();
  float den = 0.0f;
#pragma unroll
  for (int w = 0; w < NWAVES; ++w) den += redD[w];
  den *= 16.0f;
  const float invden = 1.0f / den;

  float* orow = out + (base + i) * (size_t)L_SZ;
  for (int j = tid; j < L_SZ; j += NTHREADS) {
    const bool m = row_in && (j <= i) && (j < vl);
    orow[j] = m ? erow[j] * invden : -1.0e9f;
  }
}

// ---------------------------------------------------------------------------
extern "C" void kernel_launch(void* const* d_in, const int* in_sizes, int n_in,
                              void* d_out, int out_size, void* d_ws, size_t ws_size,
                              hipStream_t stream) {
  (void)in_sizes; (void)n_in; (void)out_size; (void)ws_size;
  const float* t_s     = (const float*)d_in[0];  // (B, L) f32
  const float* s_s     = (const float*)d_in[1];  // (B, L, 2) f32
  const int*   val_len = (const int*)d_in[2];    // (B,) i32
  float*       out     = (float*)d_out;          // (B, L, L) f32

  const int N = B_SZ * L_SZ;
  // Workspace layout (~1.5 MB, fits easily in the 192 MB L2).
  float* trig          = (float*)d_ws;                 // 4*N: sinlat,coslat,sinlon,coslon
  float* inv_nt        = trig + 4 * N;                 // N
  float* inv_ns        = trig + 5 * N;                 // N
  unsigned int* rmax_b = (unsigned int*)(trig + 6 * N);// B (float bits)

  k_pre<<<(N + NTHREADS - 1) / NTHREADS, NTHREADS, 0, stream>>>(
      s_s, trig, rmax_b, N);

  dim3 grid(L_SZ, B_SZ);  // one block per (row, batch)
  k_norms_rmax<<<grid, NTHREADS, 0, stream>>>(
      t_s, val_len, trig, inv_nt, inv_ns, rmax_b);
  k_softmax_out<<<grid, NTHREADS, 0, stream>>>(
      t_s, val_len, trig, inv_nt, inv_ns, rmax_b, out);
}
